// BaiChuanAttention_40398462386177
// MI455X (gfx1250) — matrix operand order repaired and checked by reading, not compiled, not run on hardware
//
#include <hip/hip_runtime.h>

// ---------------- problem constants ----------------
constexpr int Bc  = 2;
constexpr int Sc  = 2048;
constexpr int Hc  = 4096;
constexpr int NHc = 32;
constexpr int HDc = 128;
constexpr float SCALE = 0.08838834764831845f;   // 128^-0.5
constexpr float NEG_BIG = -1.0e30f;

typedef __attribute__((ext_vector_type(16))) __bf16 v16bf;
typedef __attribute__((ext_vector_type(8)))  float  v8f;
typedef __attribute__((ext_vector_type(4)))  unsigned int v4u;
typedef __attribute__((ext_vector_type(8)))  int  v8i;
typedef __attribute__((ext_vector_type(4)))  int  v4i;

// ---------------- fragment loaders (ISA 7.12.2 layouts) ----------------
// A-operand (16x32 bf16): lane holds row m=lane%16.
//   elems 0..7  -> k = hi*8 + e          (16 bytes)
//   elems 8..15 -> k = 16 + hi*8 + (e-8) (16 bytes)
__device__ inline v16bf load_fragA(const __bf16* p, int hi) {
    v16bf f;
    ((float4*)&f)[0] = ((const float4*)(p + hi * 8))[0];
    ((float4*)&f)[1] = ((const float4*)(p + 16 + hi * 8))[0];
    return f;
}
// B-operand (32x16 bf16): lane holds column n=lane%16 (a row of the
// row-major [N,K] source). elems 0..7 -> k = hi*8 + e, elems 8..15 -> k = 16 + hi*8 + (e-8)
__device__ inline v16bf load_fragB(const __bf16* p, int hi) {
    v16bf f;
    ((float4*)&f)[0] = ((const float4*)(p + hi * 8))[0];
    ((float4*)&f)[1] = ((const float4*)(p + 16 + hi * 8))[0];
    return f;
}
__device__ inline v8f vzero8() {
    v8f z;
#pragma unroll
    for (int i = 0; i < 8; ++i) z[i] = 0.0f;
    return z;
}

// ---------------- TDM: 2-D tile load (128 rows x 32 bf16) into LDS -------
// D# per CDNA5 ISA 8.3/8.4. group0: count=1, lds_addr, 57b global addr,
// type=2. group1: data_size=2B, tensor dims, tile_dim0=32, tile_dim1=128,
// tensor_dim0_stride=K. groups 2/3 zero (2-D tensor).
__device__ inline unsigned lds_off32(const void* p) {
    return (unsigned)(uintptr_t)p;   // low 32 bits of generic ptr = LDS offset
}
__device__ inline void tdm_load_tile_128x32(unsigned lds_byte_off,
                                            const __bf16* gptr,
                                            unsigned stride_elems) {
    unsigned long long ga = (unsigned long long)(uintptr_t)gptr;
    v4u g0;
    g0[0] = 1u;                                      // count = 1 valid D#
    g0[1] = lds_byte_off;                            // lds_addr
    g0[2] = (unsigned)ga;                            // global_addr[31:0]
    g0[3] = (unsigned)((ga >> 32) & 0x01ffffffu)     // global_addr[56:32]
          | (2u << 30);                              // type = 2 ("image")
    unsigned td0 = stride_elems;                     // tensor_dim0 = K
    unsigned td1 = 128u;                             // tensor_dim1 = rows
    v8i g1;
    g1[0] = (int)(1u << 16);                         // data_size = 1 -> 2 bytes
    g1[1] = (int)((td0 & 0xffffu) << 16);            // tensor_dim0[15:0]
    g1[2] = (int)(((td0 >> 16) & 0xffffu) | ((td1 & 0xffffu) << 16));
    g1[3] = (int)(((td1 >> 16) & 0xffffu) | (32u << 16));   // tile_dim0 = 32
    g1[4] = (int)128u;                               // tile_dim1 = 128, tile_dim2=0
    g1[5] = (int)stride_elems;                       // tensor_dim0_stride[31:0]
    g1[6] = 0;                                       // stride[47:32], dim1_stride lo
    g1[7] = 0;
    v4i gz4;
    gz4[0] = 0; gz4[1] = 0; gz4[2] = 0; gz4[3] = 0;
    v8i gz8;
#pragma unroll
    for (int i = 0; i < 8; ++i) gz8[i] = 0;
    // 6-arg form on this toolchain: (g0, g1, g2, g3, g-extra, cpol)
    __builtin_amdgcn_tensor_load_to_lds(g0, g1, gz4, gz4, gz8, 0);
}

// ---------------- kernel 0: f32 -> bf16 convert ----------------
__global__ void cvt_f32_bf16(const float* __restrict__ src,
                             __bf16* __restrict__ dst, long n) {
    long i = (long)blockIdx.x * blockDim.x + threadIdx.x;
    if (i < n) dst[i] = (__bf16)src[i];
}

// ---------------- GEMM: C = A(MxK) * W(NxK)^T, TDM double-buffered -------
// block = 256 threads = 8 waves; block tile 128(M) x 128(N); wave tile
// 32(M) x 64(N) -> 8 wmma per 32-deep K-step, all operands out of LDS.
// Wave 0 drives the Tensor Data Mover; stage t+1 DMAs while stage t computes.
template <typename OutT>
__global__ __launch_bounds__(256)
void gemm_bf16_tdm(const __bf16* __restrict__ A, const __bf16* __restrict__ W,
                   OutT* __restrict__ C, int K, int N) {
    __shared__ __align__(16) __bf16 lA[2][128 * 32];
    __shared__ __align__(16) __bf16 lB[2][128 * 32];

    const int lane = threadIdx.x & 31;
    const int wv   = threadIdx.x >> 5;
    const int hi   = lane >> 4;
    const int lr   = lane & 15;
    const int wm   = wv & 3;    // 4 M-strips of 32
    const int wn   = wv >> 2;   // 2 N-strips of 64
    const int mblk = blockIdx.y * 128;
    const int nblk = blockIdx.x * 128;

    v8f acc[2][4];
#pragma unroll
    for (int mt = 0; mt < 2; ++mt)
#pragma unroll
        for (int nt = 0; nt < 4; ++nt) acc[mt][nt] = vzero8();

    const __bf16* Abase = A + (size_t)mblk * K;
    const __bf16* Wbase = W + (size_t)nblk * K;

    // prologue: stage 0 into buffer 0
    if (wv == 0) {
        tdm_load_tile_128x32(lds_off32(&lA[0][0]), Abase, (unsigned)K);
        tdm_load_tile_128x32(lds_off32(&lB[0][0]), Wbase, (unsigned)K);
    }

    const int nk = K / 32;
    for (int t = 0; t < nk; ++t) {
        const int cur = t & 1;
        if (wv == 0) {
            if (t + 1 < nk) {   // prefetch next stage into other buffer
                tdm_load_tile_128x32(lds_off32(&lA[cur ^ 1][0]),
                                     Abase + (t + 1) * 32, (unsigned)K);
                tdm_load_tile_128x32(lds_off32(&lB[cur ^ 1][0]),
                                     Wbase + (t + 1) * 32, (unsigned)K);
                __builtin_amdgcn_s_wait_tensorcnt(2);  // stage t done (in-order)
            } else {
                __builtin_amdgcn_s_wait_tensorcnt(0);
            }
        }
        __syncthreads();   // stage t visible to all waves

        const __bf16* pa = &lA[cur][0];
        const __bf16* pb = &lB[cur][0];
        v16bf af0 = load_fragA(pa + (wm * 32 + lr) * 32, hi);
        v16bf af1 = load_fragA(pa + (wm * 32 + 16 + lr) * 32, hi);
#pragma unroll
        for (int nt = 0; nt < 4; ++nt) {
            v16bf bf = load_fragB(pb + (wn * 64 + nt * 16 + lr) * 32, hi);
            acc[0][nt] = __builtin_amdgcn_wmma_f32_16x16x32_bf16(
                false, af0, false, bf, (short)0, acc[0][nt], false, false);
            acc[1][nt] = __builtin_amdgcn_wmma_f32_16x16x32_bf16(
                false, af1, false, bf, (short)0, acc[1][nt], false, false);
        }
        __syncthreads();   // all waves done with buffer `cur` before reuse
    }

    // C layout: VGPR v, lane -> (m = v + 8*hi, n = lr)
#pragma unroll
    for (int mt = 0; mt < 2; ++mt)
#pragma unroll
        for (int nt = 0; nt < 4; ++nt)
#pragma unroll
            for (int v = 0; v < 8; ++v) {
                int m = mblk + wm * 32 + mt * 16 + v + 8 * hi;
                int n = nblk + wn * 64 + nt * 16 + lr;
                C[(size_t)m * N + n] = (OutT)acc[mt][nt][v];
            }
}

// ---------------- kernel 2: RoPE + head transpose ----------------
// qkv: [B,S,3,NH,HD] bf16  ->  Q,K: [B,NH,S,HD],  Vt: [B,NH,HD,S]
__global__ void rope_transpose(const __bf16* __restrict__ qkv,
                               const int* __restrict__ pos,
                               __bf16* __restrict__ Q, __bf16* __restrict__ Kc,
                               __bf16* __restrict__ Vt) {
    long idx = (long)blockIdx.x * blockDim.x + threadIdx.x;
    if (idx >= (long)Bc * Sc * NHc * (HDc / 2)) return;
    int d = idx & 63;
    int h = (idx >> 6) & (NHc - 1);
    int s = (idx >> 11) & (Sc - 1);
    int b = (int)(idx >> 22);

    float p    = (float)pos[b * Sc + s];
    float freq = __expf(-((float)(2 * d) / (float)HDc) * __logf(10000.0f));
    float sn, cs;
    __sincosf(p * freq, &sn, &cs);

    size_t base = ((size_t)(b * Sc + s)) * (3 * Hc) + (size_t)h * HDc;
    size_t ho   = (((size_t)(b * NHc + h)) * Sc + s) * HDc;

    float x1 = (float)qkv[base + d];
    float x2 = (float)qkv[base + 64 + d];
    Q[ho + d]      = (__bf16)(x1 * cs - x2 * sn);
    Q[ho + 64 + d] = (__bf16)(x2 * cs + x1 * sn);

    x1 = (float)qkv[base + Hc + d];
    x2 = (float)qkv[base + Hc + 64 + d];
    Kc[ho + d]      = (__bf16)(x1 * cs - x2 * sn);
    Kc[ho + 64 + d] = (__bf16)(x2 * cs + x1 * sn);

    size_t vo = ((size_t)(b * NHc + h)) * HDc;
    Vt[(vo + d) * Sc + s]      = qkv[base + 2 * Hc + d];
    Vt[(vo + 64 + d) * Sc + s] = qkv[base + 2 * Hc + 64 + d];
}

// ---------------- kernel 3: causal flash attention ----------------
__global__ __launch_bounds__(128)
void flash_attn(const __bf16* __restrict__ Q, const __bf16* __restrict__ Kc,
                const __bf16* __restrict__ Vt, __bf16* __restrict__ Oout) {
    __shared__ __align__(16) __bf16 pst[4][16 * 32];   // per-wave P staging

    const int lane = threadIdx.x & 31;
    const int wv   = threadIdx.x >> 5;
    const int hi   = lane >> 4;
    const int lr   = lane & 15;
    const int qb   = blockIdx.x * 64;
    const int h    = blockIdx.y;
    const int b    = blockIdx.z;
    const int qr0  = qb + wv * 16;

    const size_t kbase  = ((size_t)(b * NHc + h)) * Sc * HDc;
    const size_t vtbase = ((size_t)(b * NHc + h)) * HDc * Sc;
    const size_t qrow   = kbase + (size_t)(qr0 + lr) * HDc;

    v16bf qf[4];
#pragma unroll
    for (int kk = 0; kk < 4; ++kk) qf[kk] = load_fragA(Q + qrow + kk * 32, hi);

    v8f o[8];
#pragma unroll
    for (int dd = 0; dd < 8; ++dd) o[dd] = vzero8();
    float mrun[8], lsum[8];
#pragma unroll
    for (int v = 0; v < 8; ++v) { mrun[v] = NEG_BIG; lsum[v] = 0.0f; }

    const int nsteps = (qb + 64) / 32;   // uniform over block (causal limit)
    __bf16* pw = pst[wv];

    for (int st = 0; st < nsteps; ++st) {
        const int j = st * 32;
        v8f s0 = vzero8(), s1 = vzero8();
#pragma unroll
        for (int kk = 0; kk < 4; ++kk) {
            v16bf k0 = load_fragB(Kc + kbase + (size_t)(j + lr) * HDc + kk * 32, hi);
            v16bf k1 = load_fragB(Kc + kbase + (size_t)(j + 16 + lr) * HDc + kk * 32, hi);
            s0 = __builtin_amdgcn_wmma_f32_16x16x32_bf16(false, qf[kk], false, k0,
                                                         (short)0, s0, false, false);
            s1 = __builtin_amdgcn_wmma_f32_16x16x32_bf16(false, qf[kk], false, k1,
                                                         (short)0, s1, false, false);
        }
#pragma unroll
        for (int v = 0; v < 8; ++v) {
            const int m  = v + 8 * hi;
            const int qg = qr0 + m;
            float a0 = (j + lr <= qg)      ? s0[v] * SCALE : NEG_BIG;
            float a1 = (j + 16 + lr <= qg) ? s1[v] * SCALE : NEG_BIG;
            float rm = fmaxf(a0, a1);
            rm = fmaxf(rm, __shfl_xor(rm, 1));
            rm = fmaxf(rm, __shfl_xor(rm, 2));
            rm = fmaxf(rm, __shfl_xor(rm, 4));
            rm = fmaxf(rm, __shfl_xor(rm, 8));
            float mn = fmaxf(mrun[v], rm);
            float al = __expf(mrun[v] - mn);
            mrun[v]  = mn;
            float p0 = __expf(a0 - mn);
            float p1 = __expf(a1 - mn);
            float rs = p0 + p1;
            rs += __shfl_xor(rs, 1);
            rs += __shfl_xor(rs, 2);
            rs += __shfl_xor(rs, 4);
            rs += __shfl_xor(rs, 8);
            lsum[v] = lsum[v] * al + rs;
#pragma unroll
            for (int dd = 0; dd < 8; ++dd) o[dd][v] *= al;
            pw[m * 32 + lr]      = (__bf16)p0;
            pw[m * 32 + 16 + lr] = (__bf16)p1;
        }
        __syncthreads();
        v16bf pf = load_fragA(pw + lr * 32, hi);
#pragma unroll
        for (int dd = 0; dd < 8; ++dd) {
            v16bf vf = load_fragB(Vt + vtbase + (size_t)(dd * 16 + lr) * Sc + j, hi);
            o[dd] = __builtin_amdgcn_wmma_f32_16x16x32_bf16(false, pf, false, vf,
                                                            (short)0, o[dd], false, false);
        }
        __syncthreads();
    }
#pragma unroll
    for (int v = 0; v < 8; ++v) {
        const int m  = v + 8 * hi;
        const int qg = qr0 + m;
        const float inv = 1.0f / lsum[v];
        size_t ob = (((size_t)(b * Sc + qg)) * NHc + h) * HDc;
#pragma unroll
        for (int dd = 0; dd < 8; ++dd)
            Oout[ob + dd * 16 + lr] = (__bf16)(o[dd][v] * inv);
    }
}

// ---------------- host-side launch ----------------
extern "C" void kernel_launch(void* const* d_in, const int* in_sizes, int n_in,
                              void* d_out, int out_size, void* d_ws, size_t ws_size,
                              hipStream_t stream) {
    const int*   positions = (const int*)d_in[0];
    const float* hidden    = (const float*)d_in[1];
    const float* w_pack    = (const float*)d_in[2];
    const float* w_o       = (const float*)d_in[3];
    float*       out       = (float*)d_out;

    constexpr size_t N_HS    = (size_t)Bc * Sc * Hc;
    constexpr size_t N_WPACK = (size_t)3 * Hc * Hc;
    constexpr size_t N_WO    = (size_t)Hc * Hc;
    constexpr size_t N_QKV   = (size_t)Bc * Sc * 3 * Hc;
    constexpr size_t N_HEAD  = (size_t)Bc * NHc * Sc * HDc;

    char* ws = (char*)d_ws;
    size_t off = 0;
    __bf16* hs_b   = (__bf16*)(ws + off); off += 2 * N_HS;
    __bf16* wp_b   = (__bf16*)(ws + off); off += 2 * N_WPACK;
    __bf16* wo_b   = (__bf16*)(ws + off); off += 2 * N_WO;
    __bf16* qkv_b  = (__bf16*)(ws + off); off += 2 * N_QKV;
    __bf16* q_b    = (__bf16*)(ws + off); off += 2 * N_HEAD;
    __bf16* k_b    = (__bf16*)(ws + off); off += 2 * N_HEAD;
    __bf16* vt_b   = (__bf16*)(ws + off); off += 2 * N_HEAD;
    __bf16* attn_b = (__bf16*)(ws + off); off += 2 * N_HEAD;

    cvt_f32_bf16<<<(unsigned)((N_HS + 255) / 256), 256, 0, stream>>>(hidden, hs_b, (long)N_HS);
    cvt_f32_bf16<<<(unsigned)((N_WPACK + 255) / 256), 256, 0, stream>>>(w_pack, wp_b, (long)N_WPACK);
    cvt_f32_bf16<<<(unsigned)((N_WO + 255) / 256), 256, 0, stream>>>(w_o, wo_b, (long)N_WO);

    // QKV projection: [4096,4096] x [12288,4096]^T -> [4096,12288] bf16
    gemm_bf16_tdm<__bf16><<<dim3((3 * Hc) / 128, (Bc * Sc) / 128), 256, 0, stream>>>(
        hs_b, wp_b, qkv_b, Hc, 3 * Hc);

    {
        long total = (long)Bc * Sc * NHc * (HDc / 2);
        rope_transpose<<<(unsigned)((total + 255) / 256), 256, 0, stream>>>(
            qkv_b, positions, q_b, k_b, vt_b);
    }

    flash_attn<<<dim3(Sc / 64, NHc, Bc), 128, 0, stream>>>(q_b, k_b, vt_b, attn_b);

    // output projection: [4096,4096] x [4096,4096]^T -> d_out f32
    gemm_bf16_tdm<float><<<dim3(Hc / 128, (Bc * Sc) / 128), 256, 0, stream>>>(
        attn_b, wo_b, out, Hc, Hc);
}
